// FDNet_38500086841533
// MI455X (gfx1250) — matrix operand order repaired and checked by reading, not compiled, not run on hardware
//
#include <hip/hip_runtime.h>
#include <hip/hip_bf16.h>

typedef __attribute__((ext_vector_type(16))) _Float16 v16h;
typedef __attribute__((ext_vector_type(8)))  _Float16 v8h;
typedef __attribute__((ext_vector_type(2)))  _Float16 v2h;
typedef __attribute__((ext_vector_type(8)))  float    v8f;

#define BN_EPS 1e-5f

// -------------------------------------------------------------------------
// Direct 3x3 conv + BN + leaky + fused 2x2 maxpool, for small Cin (3, 16).
// -------------------------------------------------------------------------
__global__ void conv_small_pool_kernel(const float* __restrict__ in,
                                       const float* __restrict__ wgt,
                                       const float* __restrict__ gamma,
                                       const float* __restrict__ beta,
                                       const float* __restrict__ mean,
                                       const float* __restrict__ var,
                                       float* __restrict__ out,
                                       int Cin, int Cout, int H, int W)
{
    const int HO = H >> 1, WO = W >> 1;
    long long tid = (long long)blockIdx.x * blockDim.x + threadIdx.x;
    long long total = 8LL * Cout * HO * WO;
    if (tid >= total) return;
    int px = (int)(tid % WO); long long t = tid / WO;
    int py = (int)(t % HO);   t /= HO;
    int oc = (int)(t % Cout); int b = (int)(t / Cout);

    float s  = gamma[oc] * rsqrtf(var[oc] + BN_EPS);
    float bb = beta[oc] - mean[oc] * s;

    const float* wo = wgt + (long long)oc * Cin * 9;
    float m = -1e30f;
#pragma unroll
    for (int dy = 0; dy < 2; ++dy) {
#pragma unroll
        for (int dx = 0; dx < 2; ++dx) {
            int yc = 2 * py + dy, xc = 2 * px + dx;
            float acc = 0.f;
            for (int ci = 0; ci < Cin; ++ci) {
                const float* ip = in + (((long long)b * Cin + ci) * H) * W;
                const float* wp = wo + ci * 9;
#pragma unroll
                for (int ky = 0; ky < 3; ++ky) {
                    int yy = yc + ky - 1;
                    if (yy < 0 || yy >= H) continue;
#pragma unroll
                    for (int kx = 0; kx < 3; ++kx) {
                        int xx = xc + kx - 1;
                        if (xx < 0 || xx >= W) continue;
                        acc += ip[yy * W + xx] * wp[ky * 3 + kx];
                    }
                }
            }
            float yv = acc * s + bb;
            yv = yv > 0.f ? yv : 0.1f * yv;
            m = fmaxf(m, yv);
        }
    }
    out[tid] = m;
}

// -------------------------------------------------------------------------
// 2x2 stride-2 maxpool, NCHW.
// -------------------------------------------------------------------------
__global__ void maxpool_kernel(const float* __restrict__ in,
                               float* __restrict__ out,
                               int C, int H, int W)
{
    int HO = H >> 1, WO = W >> 1;
    long long tid = (long long)blockIdx.x * blockDim.x + threadIdx.x;
    long long total = 8LL * C * HO * WO;
    if (tid >= total) return;
    int px = (int)(tid % WO); long long t = tid / WO;
    int py = (int)(t % HO);   t /= HO;
    int c  = (int)(t % C);    int b = (int)(t / C);
    const float* p = in + (((long long)b * C + c) * H + 2 * py) * W + 2 * px;
    out[tid] = fmaxf(fmaxf(p[0], p[1]), fmaxf(p[W], p[W + 1]));
}

// -------------------------------------------------------------------------
// Weight repack: w[Cout][Cin][kh][kw] (f32) -> wpack[oc][k'] (f16),
// k' = r*Cin + cin (tap-major), zero-padded to Kpad (multiple of 32).
// -------------------------------------------------------------------------
__global__ void repack_kernel(const float* __restrict__ w,
                              _Float16* __restrict__ wpack,
                              int Cout, int Cin, int KK, int Kpad)
{
    long long tid = (long long)blockIdx.x * blockDim.x + threadIdx.x;
    long long total = (long long)Cout * Kpad;
    if (tid >= total) return;
    int kp = (int)(tid % Kpad);
    int oc = (int)(tid / Kpad);
    float v = 0.f;
    if (kp < Cin * KK) {
        int rr  = kp / Cin;
        int cin = kp - rr * Cin;
        v = w[((long long)oc * Cin + cin) * KK + rr];
    }
    wpack[tid] = (_Float16)v;
}

// -------------------------------------------------------------------------
// WMMA implicit-GEMM conv (SAME). Block = wpb waves; all waves share one
// 16-pixel tile, each owns a 16-cout tile. Per 32-K chunk the block
// cooperatively stages the shared B fragment in LDS (f16, [pixel][k_local]),
// double-buffered, one barrier per chunk; each wave then reads two
// contiguous 16B LDS vectors and issues one v_wmma_f32_16x16x32_f16.
// A-fragment global loads are issued BEFORE the barrier so their latency
// overlaps the staging stores + barrier wait.
// K is tap-major (k = r*Cin + cin), Cin a power of two (>=32), so each
// pair's bounds predicate is uniform per chunk element pair.
// mode 0: BN+leaky  mode 1: SAM out=feat*sigmoid(acc+bias)  mode 2: +bias
// -------------------------------------------------------------------------
__global__ void wmma_conv_kernel(const float* __restrict__ in,
                                 const _Float16* __restrict__ wpack,
                                 const float* __restrict__ gamma,
                                 const float* __restrict__ beta,
                                 const float* __restrict__ mean,
                                 const float* __restrict__ var,
                                 const float* __restrict__ bias,
                                 const float* __restrict__ samfeat,
                                 float* __restrict__ out,
                                 int Cin, int cinShift, int Cout,
                                 int H, int W, int KK, int Kpad, int mode)
{
    __shared__ _Float16 lb[2][16 * 32];     // [buf][pixel*32 + k_local]

    const int tid  = threadIdx.x;
    const int lane = tid & 31;
    const int wave = tid >> 5;
    const int half = lane >> 4;
    const int l15  = lane & 15;
    const int HW   = H * W;
    const int pixb = blockIdx.x * 16;
    const int coutb = (blockIdx.y * (blockDim.x >> 5) + wave) * 16;
    const int b    = blockIdx.z;
    const int pix  = pixb + l15;

    // A row (clamped; writes masked in epilogue)
    int acout  = coutb + l15;
    int acoutc = acout < Cout ? acout : Cout - 1;
    const _Float16* wrow = wpack + (long long)acoutc * Kpad;
    const float* inb = in + (long long)b * Cin * HW;

    // Pre-compute pixel coords for the cooperative-load slots (<= 8).
    int spy[8], spx[8];
    {
        int it = 0;
        for (int t = tid; t < 256; t += blockDim.x, ++it) {
            int pp = pixb + (t >> 4);
            spy[it] = pp / W;
            spx[it] = pp - spy[it] * W;
        }
    }

    v8f acc = {};
    int nbuf = 0;
    for (int kk = 0; kk < Kpad; kk += 32, nbuf ^= 1) {
        // ---- A fragment: issue the two 16B global loads first, so their
        //      latency overlaps the LDS staging and the barrier. ----
        v8h alo = *(const v8h*)(wrow + kk + (half << 3));
        v8h ahi = *(const v8h*)(wrow + kk + 16 + (half << 3));

        // ---- cooperative B staging: 256 (pixel, channel-pair) slots ----
        int it = 0;
        for (int t = tid; t < 256; t += blockDim.x, ++it) {
            int p   = t >> 4;
            int m2  = (t & 15) << 1;            // even k_local
            int k0  = kk + m2;
            int rr  = k0 >> cinShift;           // tap (>=KK only in pad)
            int cin = k0 & (Cin - 1);
            int yy = spy[it], xx = spx[it];
            if (KK == 9) { int t3 = rr / 3; yy += t3 - 1; xx += (rr - t3 * 3) - 1; }
            bool ok = (rr < KK) & (yy >= 0) & (yy < H) & (xx >= 0) & (xx < W);
            int yyc = yy < 0 ? 0 : (yy >= H ? H - 1 : yy);
            int xxc = xx < 0 ? 0 : (xx >= W ? W - 1 : xx);
            const float* q = inb + (long long)cin * HW + yyc * W + xxc;
            float v0 = q[0];                    // unconditional loads
            float v1 = q[HW];
            v2h pr;
            pr[0] = ok ? (_Float16)v0 : (_Float16)0.f;
            pr[1] = ok ? (_Float16)v1 : (_Float16)0.f;
            *(v2h*)&lb[nbuf][p * 32 + m2] = pr;
        }
        __syncthreads();                        // stores visible; implies DS wait

        v16h a = __builtin_shufflevector(alo, ahi,
                  0,1,2,3,4,5,6,7,8,9,10,11,12,13,14,15);

        // ---- B fragment: two contiguous 16B LDS loads ----
        const v8h* lp = (const v8h*)&lb[nbuf][l15 * 32 + (half << 3)];
        v8h blo = lp[0];
        v8h bhi = lp[2];
        v16h bm = __builtin_shufflevector(blo, bhi,
                  0,1,2,3,4,5,6,7,8,9,10,11,12,13,14,15);

        acc = __builtin_amdgcn_wmma_f32_16x16x32_f16(
            false, a, false, bm, (short)0, acc, false, false);
    }

    // ---- epilogue: lane l15 = column; VGPR j = row j + 8*half ----
#pragma unroll
    for (int j = 0; j < 8; ++j) {
        int oc = coutb + j + (half << 3);
        if (oc >= Cout) continue;
        float v = acc[j];
        long long oidx = ((long long)b * Cout + oc) * HW + pix;
        if (mode == 0) {
            float s  = gamma[oc] * rsqrtf(var[oc] + BN_EPS);
            float bb = beta[oc] - mean[oc] * s;
            float y  = v * s + bb;
            out[oidx] = y > 0.f ? y : 0.1f * y;
        } else if (mode == 1) {
            float att = 1.f / (1.f + expf(-(v + bias[oc])));
            out[oidx] = samfeat[oidx] * att;
        } else {
            out[oidx] = v + bias[oc];
        }
    }
}

// -------------------------------------------------------------------------
// Decode head: pred [5][400] (batch 0) -> bbox [400][4], obj [400].
// -------------------------------------------------------------------------
__global__ void decode_kernel(const float* __restrict__ pred,
                              float* __restrict__ bbox,
                              float* __restrict__ obj)
{
    int i = blockIdx.x * blockDim.x + threadIdx.x;
    if (i >= 400) return;
    float conf = pred[i];
    float tx = pred[400 + i], ty = pred[800 + i];
    float tw = pred[1200 + i], th = pred[1600 + i];
    float gx = (float)(i % 20), gy = (float)(i / 20);
    float cx = (1.f / (1.f + expf(-tx)) + gx) * 32.f;
    float cy = (1.f / (1.f + expf(-ty)) + gy) * 32.f;
    float w = expf(tw), h = expf(th);
    float x1 = (cx - 0.5f * w) * (1.f / 640.f);
    float y1 = (cy - 0.5f * h) * (1.f / 640.f);
    float x2 = (cx + 0.5f * w) * (1.f / 640.f);
    float y2 = (cy + 0.5f * h) * (1.f / 640.f);
    bbox[i * 4 + 0] = fminf(fmaxf(x1, 0.f), 1.f);
    bbox[i * 4 + 1] = fminf(fmaxf(y1, 0.f), 1.f);
    bbox[i * 4 + 2] = fminf(fmaxf(x2, 0.f), 1.f);
    bbox[i * 4 + 3] = fminf(fmaxf(y2, 0.f), 1.f);
    obj[i] = 1.f / (1.f + expf(-conf));
}

// -------------------------------------------------------------------------
// Stable descending rank (= argsort of -key, ties by index).
// -------------------------------------------------------------------------
__global__ void order_kernel(const float* __restrict__ obj,
                             int* __restrict__ order,
                             int* __restrict__ keep_s)
{
    int i = blockIdx.x * blockDim.x + threadIdx.x;
    if (i >= 400) return;
    float oi = obj[i];
    float ki = (oi >= 0.01f) ? oi : -1.0f;
    int rank = 0;
    for (int j = 0; j < 400; ++j) {
        float oj = obj[j];
        float kj = (oj >= 0.01f) ? oj : -1.0f;
        if (kj > ki || (kj == ki && j < i)) ++rank;
    }
    order[rank] = i;
    keep_s[rank] = (oi >= 0.01f) ? 1 : 0;
}

// -------------------------------------------------------------------------
// IoU matrix in sorted index space.
// -------------------------------------------------------------------------
__global__ void iou_kernel(const float* __restrict__ bbox,
                           const int* __restrict__ order,
                           float* __restrict__ iou)
{
    int t = blockIdx.x * blockDim.x + threadIdx.x;
    if (t >= 400 * 400) return;
    int i = t / 400, j = t % 400;
    const float* bi = bbox + order[i] * 4;
    const float* bj = bbox + order[j] * 4;
    float xx1 = fmaxf(bi[0], bj[0]);
    float yy1 = fmaxf(bi[1], bj[1]);
    float xx2 = fminf(bi[2], bj[2]);
    float yy2 = fminf(bi[3], bj[3]);
    float inter = fmaxf(1e-28f, xx2 - xx1) * fmaxf(1e-28f, yy2 - yy1);
    float ai = (bi[2] - bi[0]) * (bi[3] - bi[1]);
    float aj = (bj[2] - bj[0]) * (bj[3] - bj[1]);
    iou[t] = inter / (ai + aj - inter);
}

// -------------------------------------------------------------------------
// Sequential greedy suppression (single workgroup, matches fori_loop).
// -------------------------------------------------------------------------
__global__ void suppress_kernel(const float* __restrict__ iou,
                                int* __restrict__ keep_s)
{
    __shared__ int k[400];
    for (int i = threadIdx.x; i < 400; i += blockDim.x) k[i] = keep_s[i];
    __syncthreads();
    for (int i = 0; i < 400; ++i) {
        if (k[i]) {
            for (int j = threadIdx.x; j < 400; j += blockDim.x)
                if (j > i && iou[i * 400 + j] > 0.3f) k[j] = 0;
        }
        __syncthreads();
    }
    for (int i = threadIdx.x; i < 400; i += blockDim.x) keep_s[i] = k[i];
}

// -------------------------------------------------------------------------
// Final scatter: bbox*m (1600), obj*m (400), keep (400), all float.
// -------------------------------------------------------------------------
__global__ void finalize_kernel(const float* __restrict__ bbox,
                                const float* __restrict__ obj,
                                const int* __restrict__ order,
                                const int* __restrict__ keep_s,
                                float* __restrict__ out)
{
    int pos = blockIdx.x * blockDim.x + threadIdx.x;
    if (pos >= 400) return;
    int orig = order[pos];
    float m = keep_s[pos] ? 1.f : 0.f;
    out[orig * 4 + 0] = bbox[orig * 4 + 0] * m;
    out[orig * 4 + 1] = bbox[orig * 4 + 1] * m;
    out[orig * 4 + 2] = bbox[orig * 4 + 2] * m;
    out[orig * 4 + 3] = bbox[orig * 4 + 3] * m;
    out[1600 + orig] = obj[orig] * m;
    out[2000 + orig] = m;
}

// -------------------------------------------------------------------------
// Host orchestration
// -------------------------------------------------------------------------
extern "C" void kernel_launch(void* const* d_in, const int* in_sizes, int n_in,
                              void* d_out, int out_size, void* d_ws, size_t ws_size,
                              hipStream_t stream)
{
    // setup_inputs() insertion order:
    // 0: x ; 1..30: backbone[i]{w,gamma,beta,mean,var} ; 31: sam.w 32: sam.b ;
    // 33..42: conv_set[i]{w,gamma,beta,mean,var} ; 43: pred.w 44: pred.b
    const float* x = (const float*)d_in[0];
    const float *bw[6], *bg[6], *bb[6], *bm[6], *bv[6];
    int idx = 1;
    for (int i = 0; i < 6; ++i) {
        bw[i] = (const float*)d_in[idx++]; bg[i] = (const float*)d_in[idx++];
        bb[i] = (const float*)d_in[idx++]; bm[i] = (const float*)d_in[idx++];
        bv[i] = (const float*)d_in[idx++];
    }
    const float* samw = (const float*)d_in[idx++];
    const float* samb = (const float*)d_in[idx++];
    const float *cw[2], *cg[2], *cb[2], *cm[2], *cv[2];
    for (int i = 0; i < 2; ++i) {
        cw[i] = (const float*)d_in[idx++]; cg[i] = (const float*)d_in[idx++];
        cb[i] = (const float*)d_in[idx++]; cm[i] = (const float*)d_in[idx++];
        cv[i] = (const float*)d_in[idx++];
    }
    const float* pw = (const float*)d_in[idx++];
    const float* pb = (const float*)d_in[idx++];

    // Workspace layout
    char* ws = (char*)d_ws;
    float* bufA = (float*)ws;                                // <= 26 MB
    float* bufB = (float*)(ws + 26214400);                   // <= 52 MB
    _Float16* wpack = (_Float16*)(ws + 26214400 + 52428800); // <= 4.7 MB
    char* sm = ws + 26214400 + 52428800 + 4718592;
    float* predb  = (float*)sm; sm += 16 * 400 * 4;
    float* bboxb  = (float*)sm; sm += 1600 * 4;
    float* objb   = (float*)sm; sm += 400 * 4;
    int*   orderb = (int*)sm;   sm += 400 * 4;
    int*   keepb  = (int*)sm;   sm += 400 * 4;
    float* ioub   = (float*)sm;

    auto conv = [&](const float* in, const float* w, const float* g,
                    const float* be, const float* me, const float* va,
                    const float* bias, const float* sf, float* out,
                    int Cin, int Cout, int H, int W, int ks, int mode, int batch) {
        int KK = ks * ks;
        int Kpad = ((Cin * KK + 31) / 32) * 32;
        int cinShift = __builtin_ctz((unsigned)Cin);
        long long rtotal = (long long)Cout * Kpad;
        repack_kernel<<<(unsigned)((rtotal + 255) / 256), 256, 0, stream>>>(
            w, wpack, Cout, Cin, KK, Kpad);
        int ctiles = (Cout + 15) / 16;
        int wpb = ctiles < 8 ? ctiles : 8;           // waves per block
        dim3 grid((H * W) / 16, (ctiles + wpb - 1) / wpb, batch);
        wmma_conv_kernel<<<grid, 32 * wpb, 0, stream>>>(
            in, wpack, g, be, me, va, bias, sf, out,
            Cin, cinShift, Cout, H, W, KK, Kpad, mode);
    };
    auto pool = [&](const float* in, float* out, int C, int H, int W) {
        long long total = 8LL * C * (H / 2) * (W / 2);
        maxpool_kernel<<<(unsigned)((total + 255) / 256), 256, 0, stream>>>(in, out, C, H, W);
    };

    // Layer 1 (direct, fused pool): x -> bufB [8,16,320,320]
    {
        long long total = 8LL * 16 * 320 * 320;
        conv_small_pool_kernel<<<(unsigned)((total + 255) / 256), 256, 0, stream>>>(
            x, bw[0], bg[0], bb[0], bm[0], bv[0], bufB, 3, 16, 640, 640);
    }
    // Layer 2 (direct, fused pool): bufB -> bufA [8,32,160,160]
    {
        long long total = 8LL * 32 * 160 * 160;
        conv_small_pool_kernel<<<(unsigned)((total + 255) / 256), 256, 0, stream>>>(
            bufB, bw[1], bg[1], bb[1], bm[1], bv[1], bufA, 16, 32, 320, 320);
    }
    // Layers 3..6 (WMMA) with pools
    conv(bufA, bw[2], bg[2], bb[2], bm[2], bv[2], nullptr, nullptr, bufB, 32, 64, 160, 160, 3, 0, 8);
    pool(bufB, bufA, 64, 160, 160);
    conv(bufA, bw[3], bg[3], bb[3], bm[3], bv[3], nullptr, nullptr, bufB, 64, 128, 80, 80, 3, 0, 8);
    pool(bufB, bufA, 128, 80, 80);
    conv(bufA, bw[4], bg[4], bb[4], bm[4], bv[4], nullptr, nullptr, bufB, 128, 256, 40, 40, 3, 0, 8);
    pool(bufB, bufA, 256, 40, 40);
    conv(bufA, bw[5], bg[5], bb[5], bm[5], bv[5], nullptr, nullptr, bufB, 256, 512, 20, 20, 3, 0, 8);
    // SAM: bufA = bufB * sigmoid(conv1x1(bufB) + samb)
    conv(bufB, samw, nullptr, nullptr, nullptr, nullptr, samb, bufB, bufA, 512, 512, 20, 20, 1, 1, 8);
    // conv_set
    conv(bufA, cw[0], cg[0], cb[0], cm[0], cv[0], nullptr, nullptr, bufB, 512, 512, 20, 20, 3, 0, 8);
    conv(bufB, cw[1], cg[1], cb[1], cm[1], cv[1], nullptr, nullptr, bufA, 512, 512, 20, 20, 3, 0, 8);
    // pred head (batch 0 only)
    conv(bufA, pw, nullptr, nullptr, nullptr, nullptr, pb, nullptr, predb, 512, 5, 20, 20, 1, 2, 1);

    // Decode + NMS
    decode_kernel<<<2, 256, 0, stream>>>(predb, bboxb, objb);
    order_kernel<<<2, 256, 0, stream>>>(objb, orderb, keepb);
    iou_kernel<<<(400 * 400 + 255) / 256, 256, 0, stream>>>(bboxb, orderb, ioub);
    suppress_kernel<<<1, 512, 0, stream>>>(ioub, keepb);
    finalize_kernel<<<2, 256, 0, stream>>>(bboxb, objb, orderb, keepb, (float*)d_out);

    (void)in_sizes; (void)n_in; (void)out_size; (void)ws_size;
}
